// Attention_61753039782117
// MI455X (gfx1250) — matrix operand order repaired and checked by reading, not compiled, not run on hardware
//
#include <hip/hip_runtime.h>
#include <hip/hip_bf16.h>

#define SEQ   2048
#define EMB   1024
#define HEADS 16
#define HD    64
#define BATCH 2
#define PST   40   // LDS row stride (ushorts) for P tile; 80B rows keep 16B alignment

typedef __attribute__((ext_vector_type(16))) __bf16          v16bf;
typedef __attribute__((ext_vector_type(16))) unsigned short  v16u;
typedef __attribute__((ext_vector_type(8)))  float           v8f;
typedef __attribute__((ext_vector_type(4)))  unsigned int    v4u;
typedef __attribute__((ext_vector_type(8)))  int             v8i;
typedef __attribute__((ext_vector_type(4)))  int             v4i;

#if defined(__HIP_DEVICE_COMPILE__) && defined(__has_builtin)
#if __has_builtin(__builtin_amdgcn_tensor_load_to_lds) && __has_builtin(__builtin_amdgcn_s_wait_tensorcnt)
#define HAVE_TDM 1
#endif
#if __has_builtin(__builtin_amdgcn_permlane16)
#define HAVE_PERMLANE 1
#define PERMLANE16(o, s, a, b) __builtin_amdgcn_permlane16((o), (s), (a), (b), false, false)
#elif __has_builtin(__builtin_amdgcn_permlane16_b32)
#define HAVE_PERMLANE 1
#define PERMLANE16(o, s, a, b) __builtin_amdgcn_permlane16_b32((o), (s), (a), (b), false, false)
#endif
#endif
#ifndef HAVE_TDM
#define HAVE_TDM 0
#endif
#ifndef HAVE_PERMLANE
#define HAVE_PERMLANE 0
#endif

union ABf {
  v16u  u;
  v16bf b;
  uint4 q[2];
};

__device__ __forceinline__ unsigned short f2bf(float f) {
  unsigned int u = __float_as_uint(f);
  u += 0x7FFFu + ((u >> 16) & 1u);
  return (unsigned short)(u >> 16);
}

// Pack two f32 -> dword of two bf16 with one v_perm_b32.
__device__ __forceinline__ unsigned pack_bf16x2(float lo, float hi) {
  unsigned ulo = __float_as_uint(lo) + 0x8000u;
  unsigned uhi = __float_as_uint(hi) + 0x8000u;
  return __builtin_amdgcn_perm(uhi, ulo, 0x07060302u);
}

__device__ __forceinline__ v8f wmma_bf16(v16bf a, v16bf b, v8f c) {
  return __builtin_amdgcn_wmma_f32_16x16x32_bf16(false, a, false, b, (short)0, c, false, false);
}

// A-fragment (16x32 bf16): lane (m=lane&15, h=lane>>4), k = 16h + e (16 contiguous)
__device__ __forceinline__ v16bf load_a_bf16(const unsigned short* row, int h) {
  ABf fr;
  fr.q[0] = *(const uint4*)(row + h * 16);
  fr.q[1] = *(const uint4*)(row + h * 16 + 8);
  return fr.b;
}

// B-fragment (32x16 bf16): lane (n=lane&15, h=lane>>4), k = 16h + e (16 contiguous).
__device__ __forceinline__ v16bf load_b_bf16(const unsigned short* p) {
  ABf fr;
  fr.q[0] = ((const uint4*)p)[0];
  fr.q[1] = ((const uint4*)p)[1];
  return fr.b;
}

// ---- Cross-lane reductions over 16-lane rows -------------------------------
// v_permlane16_b32 selector nibbles = lane ^ mask (all masks < 16 stay in-row),
// so the whole butterfly runs in the VALU: no ds_bpermute, no DScnt waits.
#if HAVE_PERMLANE
__device__ __forceinline__ float lane_xor16(float v, unsigned s0, unsigned s1) {
  unsigned u = __float_as_uint(v);
  unsigned r = (unsigned)PERMLANE16((int)u, (int)u, (int)s0, (int)s1);
  return __uint_as_float(r);
}
__device__ __forceinline__ float rmax16(float v) {
  v = fmaxf(v, lane_xor16(v, 0x67452301u, 0xEFCDAB89u));  // xor 1
  v = fmaxf(v, lane_xor16(v, 0x54761032u, 0xDCFE98BAu));  // xor 2
  v = fmaxf(v, lane_xor16(v, 0x32107654u, 0xBA98FEDCu));  // xor 4
  v = fmaxf(v, lane_xor16(v, 0xFEDCBA98u, 0x76543210u));  // xor 8
  return v;
}
__device__ __forceinline__ float rsum16(float v) {
  v += lane_xor16(v, 0x67452301u, 0xEFCDAB89u);
  v += lane_xor16(v, 0x54761032u, 0xDCFE98BAu);
  v += lane_xor16(v, 0x32107654u, 0xBA98FEDCu);
  v += lane_xor16(v, 0xFEDCBA98u, 0x76543210u);
  return v;
}
#else
__device__ __forceinline__ float rmax16(float v) {
#pragma unroll
  for (int m = 1; m < 16; m <<= 1) v = fmaxf(v, __shfl_xor(v, m, 32));
  return v;
}
__device__ __forceinline__ float rsum16(float v) {
#pragma unroll
  for (int m = 1; m < 16; m <<= 1) v += __shfl_xor(v, m, 32);
  return v;
}
#endif

#if HAVE_TDM
// ---- Tensor DMA descriptor packing (CDNA5 ISA) ----
__device__ __forceinline__ v4u tdm_g0(unsigned lds_byte, unsigned long long ga) {
  v4u g;
  g[0] = 1u;                                                // count=1, user D#
  g[1] = lds_byte;                                          // LDS byte address
  g[2] = (unsigned)ga;                                      // global addr lo
  g[3] = (unsigned)((ga >> 32) & 0x01FFFFFFull) | (2u << 30);  // addr hi | type=2
  return g;
}
__device__ __forceinline__ v8i tdm_g1_2d(unsigned tensor_d0, unsigned tensor_d1,
                                         unsigned tile_d0, unsigned tile_d1,
                                         unsigned stride0) {
  v8i g;
  g[0] = (int)(1u << 16);                                   // data_size = 2B
  g[1] = (int)((tensor_d0 & 0xFFFFu) << 16);
  g[2] = (int)(((tensor_d0 >> 16) & 0xFFFFu) | ((tensor_d1 & 0xFFFFu) << 16));
  g[3] = (int)(((tensor_d1 >> 16) & 0xFFFFu) | ((tile_d0 & 0xFFFFu) << 16));
  g[4] = (int)(tile_d1 & 0xFFFFu);                          // tile_dim2 = 0
  g[5] = (int)stride0;
  g[6] = 0;
  g[7] = 0;
  return g;
}
__device__ __forceinline__ void tdm_load_2d(unsigned lds_byte, const void* gptr, v8i g1) {
  v4i z4 = (v4i){0, 0, 0, 0};
  v8i z8 = (v8i){0, 0, 0, 0, 0, 0, 0, 0};
  __builtin_amdgcn_tensor_load_to_lds(tdm_g0(lds_byte, (unsigned long long)gptr),
                                      g1, z4, z4, z8, 0);
}
#endif

// ---------------------------------------------------------------------------
// One-shot f32 -> bf16 conversion.
// ---------------------------------------------------------------------------
__global__ __launch_bounds__(256) void cvt_bf16_kernel(
    const float* __restrict__ src, unsigned short* __restrict__ dst, int n8) {
  int i = blockIdx.x * 256 + threadIdx.x;
  if (i >= n8) return;
  const float4* s = (const float4*)src;
  float4 x0 = s[2 * i], x1 = s[2 * i + 1];
  uint4 o;
  o.x = pack_bf16x2(x0.x, x0.y);
  o.y = pack_bf16x2(x0.z, x0.w);
  o.z = pack_bf16x2(x1.x, x1.y);
  o.w = pack_bf16x2(x1.z, x1.w);
  ((uint4*)dst)[i] = o;
}

// ---------------------------------------------------------------------------
// GEMM: out = A(bf16)[M,EMB] @ W(bf16)[N,EMB]^T + bias. 32x64 tile per wave.
// MODE 0: bf16 out [B,H,SEQ,HD]; MODE 1: bf16 out [B,H,HD,SEQ]; MODE 2: f32 [M,EMB].
// ---------------------------------------------------------------------------
template <int MODE>
__global__ __launch_bounds__(128) void gemm_bf16_kernel(
    const unsigned short* __restrict__ A, const unsigned short* __restrict__ W,
    const float* __restrict__ bias, unsigned short* __restrict__ outb,
    float* __restrict__ outf) {
  int lane = threadIdx.x & 31;
  int t  = blockIdx.x * 4 + (threadIdx.x >> 5);
  int tm = t >> 4;
  int tn = t & 15;
  int c = lane & 15, hh = lane >> 4;

  const unsigned short* A0 = A + (size_t)(tm * 32 + c) * EMB;
  const unsigned short* A1 = A0 + (size_t)16 * EMB;
  const unsigned short* W0 = W + (size_t)(tn * 64 + c) * EMB + hh * 16;

  v8f acc[2][4];
#pragma unroll
  for (int i = 0; i < 2; ++i)
#pragma unroll
    for (int j = 0; j < 4; ++j) acc[i][j] = (v8f){};

  for (int k0 = 0; k0 < EMB; k0 += 32) {
    v16bf a0 = load_a_bf16(A0 + k0, hh);
    v16bf a1 = load_a_bf16(A1 + k0, hh);
#pragma unroll
    for (int j = 0; j < 4; ++j) {
      v16bf bf = load_b_bf16(W0 + (size_t)j * 16 * EMB + k0);
      acc[0][j] = wmma_bf16(a0, bf, acc[0][j]);
      acc[1][j] = wmma_bf16(a1, bf, acc[1][j]);
    }
  }

#pragma unroll
  for (int j = 0; j < 4; ++j) {
    int nglob = tn * 64 + j * 16 + c;
    float bv = bias[nglob];
    int head = nglob >> 6, d = nglob & (HD - 1);
#pragma unroll
    for (int i = 0; i < 2; ++i) {
#pragma unroll
      for (int r = 0; r < 8; ++r) {
        int mglob = tm * 32 + i * 16 + r + 8 * hh;
        float val = acc[i][j][r] + bv;
        if (MODE == 2) {
          outf[(size_t)mglob * EMB + nglob] = val;
        } else {
          int b = mglob >> 11, seq = mglob & (SEQ - 1);
          size_t idx;
          if (MODE == 1)
            idx = ((size_t)(b * HEADS + head) * HD + d) * SEQ + seq;
          else
            idx = ((size_t)(b * HEADS + head) * SEQ + seq) * HD + d;
          outb[idx] = f2bf(val);
        }
      }
    }
  }
}

// ---------------------------------------------------------------------------
// Flash attention, 2 waves/block, TDM-staged K/V tiles (double-buffered,
// TENSORcnt-synchronized), permlane16 softmax reductions, LDS P-bounce.
// ---------------------------------------------------------------------------
struct WaveLds {
  unsigned short K[2][32 * HD];   // 32 keys x 64 d, row stride 64  (8 KB)
  unsigned short V[2][HD * 32];   // 64 d   x 32 keys, row stride 32 (8 KB)
  unsigned short P[16 * PST];     // P tile (1.25 KB)
};

__global__ __launch_bounds__(64) void attn_kernel(
    const unsigned short* __restrict__ Qh, const unsigned short* __restrict__ Kh,
    const unsigned short* __restrict__ Vt, unsigned short* __restrict__ Ao) {
  __shared__ WaveLds lds[2];

  int lane = threadIdx.x & 31;
  int wv   = threadIdx.x >> 5;
  int w    = blockIdx.x * 2 + wv;
  int bh   = w >> 7;
  int qt   = w & 127;
  int q0   = qt * 16;
  int c = lane & 15, hh = lane >> 4;

  const unsigned short* Qb = Qh + (size_t)bh * SEQ * HD;
  const unsigned short* Kb = Kh + (size_t)bh * SEQ * HD;
  const unsigned short* Vb = Vt + (size_t)bh * HD * SEQ;
  WaveLds* S = &lds[wv];

  const unsigned short* Qrow = Qb + (size_t)(q0 + c) * HD;
  v16bf qf0 = load_a_bf16(Qrow, hh);
  v16bf qf1 = load_a_bf16(Qrow + 32, hh);

  v8f o0 = {}, o1 = {}, o2 = {}, o3 = {};
  float mrun[8], lrun[8];
#pragma unroll
  for (int r = 0; r < 8; ++r) { mrun[r] = -1e30f; lrun[r] = 0.0f; }
  const float scale = 0.125f;

#if HAVE_TDM
  char* base0 = (char*)&lds[0];
  unsigned kOff[2] = { (unsigned)((char*)S->K[0] - base0), (unsigned)((char*)S->K[1] - base0) };
  unsigned vOff[2] = { (unsigned)((char*)S->V[0] - base0), (unsigned)((char*)S->V[1] - base0) };
  v8i g1K = tdm_g1_2d(HD,  SEQ, HD, 32, HD);
  v8i g1V = tdm_g1_2d(SEQ, HD,  32, HD, SEQ);
  tdm_load_2d(kOff[0], Kb, g1K);
  tdm_load_2d(vOff[0], Vb, g1V);
#endif

  for (int ch = 0; ch < SEQ / 32; ++ch) {
    int k0 = ch * 32;
    int buf = ch & 1;

#if HAVE_TDM
    asm volatile("s_wait_dscnt 0x0" ::: "memory");   // LDS reads retire before TDM overwrite
    int nch = (ch + 1) & (SEQ / 32 - 1);
    tdm_load_2d(kOff[buf ^ 1], Kb + (size_t)nch * 32 * HD, g1K);
    tdm_load_2d(vOff[buf ^ 1], Vb + (size_t)nch * 32, g1V);
    __builtin_amdgcn_s_wait_tensorcnt(2);            // chunk ch's loads landed
    asm volatile("" ::: "memory");
    const unsigned short* Kbase = S->K[buf];
    const unsigned short* Vbase = S->V[buf];
    const int vstride = 32;
    const int kv0 = 0;
#else
    const unsigned short* Kbase = Kb + (size_t)k0 * HD;
    const unsigned short* Vbase = Vb;
    const int vstride = SEQ;
    const int kv0 = k0;
    if (k0 + 32 < SEQ) {
      __builtin_prefetch(Kbase + (size_t)(32 + c) * HD, 0, 0);
      __builtin_prefetch(Vbase + (size_t)c * SEQ + k0 + 32, 0, 0);
    }
#endif

    v8f s0 = {}, s1 = {};
    const unsigned short* K0 = Kbase + (size_t)c * HD + hh * 16;
    const unsigned short* K1 = Kbase + (size_t)(16 + c) * HD + hh * 16;
    s0 = wmma_bf16(qf0, load_b_bf16(K0), s0);
    s1 = wmma_bf16(qf0, load_b_bf16(K1), s1);
    s0 = wmma_bf16(qf1, load_b_bf16(K0 + 32), s0);
    s1 = wmma_bf16(qf1, load_b_bf16(K1 + 32), s1);

#pragma unroll
    for (int r = 0; r < 8; ++r) {
      float a0 = s0[r] * scale;
      float a1 = s1[r] * scale;
      float mx    = rmax16(fmaxf(a0, a1));
      float mnew  = fmaxf(mrun[r], mx);
      float alpha = __expf(mrun[r] - mnew);
      float e0 = __expf(a0 - mnew);
      float e1 = __expf(a1 - mnew);
      float sum = rsum16(e0 + e1);
      lrun[r] = lrun[r] * alpha + sum;
      mrun[r] = mnew;
      o0[r] *= alpha; o1[r] *= alpha; o2[r] *= alpha; o3[r] *= alpha;
      int row = r + 8 * hh;
      S->P[row * PST + c]      = f2bf(e0);
      S->P[row * PST + 16 + c] = f2bf(e1);
    }
    asm volatile("s_wait_dscnt 0x0" ::: "memory");

    v16bf ap = load_a_bf16(S->P + c * PST, hh);
    o0 = wmma_bf16(ap, load_b_bf16(Vbase + (size_t)(0 * 16 + c) * vstride + kv0 + hh * 16), o0);
    o1 = wmma_bf16(ap, load_b_bf16(Vbase + (size_t)(1 * 16 + c) * vstride + kv0 + hh * 16), o1);
    o2 = wmma_bf16(ap, load_b_bf16(Vbase + (size_t)(2 * 16 + c) * vstride + kv0 + hh * 16), o2);
    o3 = wmma_bf16(ap, load_b_bf16(Vbase + (size_t)(3 * 16 + c) * vstride + kv0 + hh * 16), o3);
  }

  int b = bh >> 4, h = bh & 15;
#pragma unroll
  for (int r = 0; r < 8; ++r) {
    float inv = 1.0f / lrun[r];
    int row = q0 + r + 8 * hh;
    size_t base = ((size_t)(b * SEQ + row)) * EMB + h * HD;
    Ao[base + 0 * 16 + c] = f2bf(o0[r] * inv);
    Ao[base + 1 * 16 + c] = f2bf(o1[r] * inv);
    Ao[base + 2 * 16 + c] = f2bf(o2[r] * inv);
    Ao[base + 3 * 16 + c] = f2bf(o3[r] * inv);
  }
}

extern "C" void kernel_launch(void* const* d_in, const int* in_sizes, int n_in,
                              void* d_out, int out_size, void* d_ws, size_t ws_size,
                              hipStream_t stream) {
  (void)in_sizes; (void)n_in; (void)out_size; (void)ws_size;
  const float* q  = (const float*)d_in[0];
  const float* k  = (const float*)d_in[1];
  const float* v  = (const float*)d_in[2];
  const float* Wq = (const float*)d_in[3];
  const float* bq = (const float*)d_in[4];
  const float* Wk = (const float*)d_in[5];
  const float* bk = (const float*)d_in[6];
  const float* Wv = (const float*)d_in[7];
  const float* bv = (const float*)d_in[8];
  const float* Wo = (const float*)d_in[9];
  const float* bo = (const float*)d_in[10];
  float* out = (float*)d_out;

  const size_t AC = (size_t)BATCH * SEQ * EMB;   // 4M bf16
  const size_t WC = (size_t)EMB * EMB;           // 1M bf16
  unsigned short* qb  = (unsigned short*)d_ws;
  unsigned short* kb  = qb  + AC;
  unsigned short* vb  = kb  + AC;
  unsigned short* Wqb = vb  + AC;
  unsigned short* Wkb = Wqb + WC;
  unsigned short* Wvb = Wkb + WC;
  unsigned short* Wob = Wvb + WC;
  unsigned short* Qh  = Wob + WC;
  unsigned short* Kh  = Qh  + AC;
  unsigned short* Vt  = Kh  + AC;
  unsigned short* Ao  = Vt  + AC;       // 64 MB total

  int n8a = (int)(AC / 8), n8w = (int)(WC / 8);
  dim3 cb(256);
  cvt_bf16_kernel<<<dim3((n8a + 255) / 256), cb, 0, stream>>>(q,  qb,  n8a);
  cvt_bf16_kernel<<<dim3((n8a + 255) / 256), cb, 0, stream>>>(k,  kb,  n8a);
  cvt_bf16_kernel<<<dim3((n8a + 255) / 256), cb, 0, stream>>>(v,  vb,  n8a);
  cvt_bf16_kernel<<<dim3((n8w + 255) / 256), cb, 0, stream>>>(Wq, Wqb, n8w);
  cvt_bf16_kernel<<<dim3((n8w + 255) / 256), cb, 0, stream>>>(Wk, Wkb, n8w);
  cvt_bf16_kernel<<<dim3((n8w + 255) / 256), cb, 0, stream>>>(Wv, Wvb, n8w);
  cvt_bf16_kernel<<<dim3((n8w + 255) / 256), cb, 0, stream>>>(Wo, Wob, n8w);

  dim3 blk(128), ggemm(512);
  gemm_bf16_kernel<0><<<ggemm, blk, 0, stream>>>(qb, Wqb, bq, Qh, nullptr);
  gemm_bf16_kernel<0><<<ggemm, blk, 0, stream>>>(kb, Wkb, bk, Kh, nullptr);
  gemm_bf16_kernel<1><<<ggemm, blk, 0, stream>>>(vb, Wvb, bv, Vt, nullptr);

  attn_kernel<<<dim3(2048), dim3(64), 0, stream>>>(Qh, Kh, Vt, Ao);

  gemm_bf16_kernel<2><<<ggemm, blk, 0, stream>>>(Ao, Wob, bo, nullptr, out);
}
